// TaskPlacementGNN_15616501088439
// MI455X (gfx1250) — matrix operand order repaired and checked by reading, not compiled, not run on hardware
//
#include <hip/hip_runtime.h>
#include <hip/hip_bf16.h>

// ---------------------------------------------------------------------------
// TaskPlacementGNN on MI455X (gfx1250, wave32, WMMA f16 16x16x32)
// ---------------------------------------------------------------------------
#define NT   25000          // tasks
#define NP   512            // platforms
#define NN   25512          // nodes
#define NE   800000         // edges
#define KE   160            // edge-MLP K (131 padded to 5*32) in global layout
#define KEP  168            // edge-MLP K row stride in LDS (bank-conflict pad)
#define LPAD 8              // LDS row pad (halves, keeps 16B alignment)

typedef __attribute__((ext_vector_type(16))) _Float16 v16h;
typedef __attribute__((ext_vector_type(8)))  _Float16 h8;
typedef __attribute__((ext_vector_type(8)))  float    v8f;

__device__ __forceinline__ v8f wmma_f16(v16h a, v16h b, v8f c) {
    return __builtin_amdgcn_wmma_f32_16x16x32_f16(
        false, a, false, b, (short)0, c, false, false);
}

// A-fragment (16x32 f16): lane holds row M=lane&15,
// halves 0..7 = K kb..kb+7, halves 8..15 = K kb+16..kb+23, kb = kbase+(lane>>4)*8
__device__ __forceinline__ v16h frag_a(const _Float16* rowp, int kbase, int lane) {
    int kb = kbase + ((lane >> 4) << 3);
    h8 lo = *(const h8*)(rowp + kb);
    h8 hi = *(const h8*)(rowp + kb + 16);
    return __builtin_shufflevector(lo, hi, 0,1,2,3,4,5,6,7,8,9,10,11,12,13,14,15);
}

// B-fragment (32x16 f16), weights transposed [N][K]:
// lane holds col N=lane&15, halves 0..7 = K kb..kb+7, halves 8..15 = K kb+16..kb+23, kb = kbase+(lane>>4)*8
__device__ __forceinline__ v16h frag_b(const _Float16* rowp, int kbase, int lane) {
    int kb = kbase + ((lane >> 4) << 3);
    h8 lo = *(const h8*)(rowp + kb);
    h8 hi = *(const h8*)(rowp + kb + 16);
    return __builtin_shufflevector(lo, hi, 0,1,2,3,4,5,6,7,8,9,10,11,12,13,14,15);
}

#define WAVE_DS_FENCE() asm volatile("s_wait_dscnt 0x0" ::: "memory")

// stage [N][K] (halves, K mult of 8) global -> LDS with row stride K+LPAD
__device__ __forceinline__ void stage_w(const _Float16* __restrict__ g,
                                        _Float16* l, int N, int K, int tid, int nthr) {
    int per = K >> 3;                    // int4 chunks per row
    int tot = N * per;
    for (int i = tid; i < tot; i += nthr) {
        int n  = i / per;
        int kc = (i - n * per) << 3;
        *(int4*)(l + n * (K + LPAD) + kc) = *(const int4*)(g + n * K + kc);
    }
}

// ---------------------------------------------------------------------------
// Weight prep: fp32 [K][N] -> f16 transposed [N][Kpad] (zero-padded K)
// ---------------------------------------------------------------------------
__global__ void wprep_kernel(const float* __restrict__ src, _Float16* __restrict__ dst,
                             int K, int N, int Kpad) {
    int i = blockIdx.x * 256 + threadIdx.x;
    if (i < N * Kpad) {
        int n = i / Kpad;
        int k = i - n * Kpad;
        dst[i] = (k < K) ? (_Float16)src[k * N + n] : (_Float16)0.f;
    }
}

// ---------------------------------------------------------------------------
// Node encoders: linear(din->128) -> layernorm -> relu -> linear(128->64)
// One wave per node; 8 waves/block; 3189 blocks * 8 = 25512 exactly.
// ---------------------------------------------------------------------------
__global__ void encoder_kernel(
    const float* __restrict__ tf, const float* __restrict__ pf,
    const float* __restrict__ tw1, const float* __restrict__ tb1,
    const float* __restrict__ tg,  const float* __restrict__ tbe,
    const float* __restrict__ tw2, const float* __restrict__ tb2,
    const float* __restrict__ pw1, const float* __restrict__ pb1,
    const float* __restrict__ pg,  const float* __restrict__ pbe,
    const float* __restrict__ pw2, const float* __restrict__ pb2,
    float* __restrict__ xout)
{
    const int lane = threadIdx.x & 31;
    const int w    = threadIdx.x >> 5;
    const int n    = blockIdx.x * 8 + w;
    __shared__ float hbuf[8][128];

    const float *xin, *W1, *B1, *G, *BE, *W2, *B2;
    int din;
    if (n < NT) { xin = tf + (size_t)n * 3;        din = 3; W1=tw1;B1=tb1;G=tg;BE=tbe;W2=tw2;B2=tb2; }
    else        { xin = pf + (size_t)(n - NT) * 7; din = 7; W1=pw1;B1=pb1;G=pg;BE=pbe;W2=pw2;B2=pb2; }

    float h[4];
    #pragma unroll
    for (int j = 0; j < 4; ++j) {
        int u = lane + 32 * j;
        float a = B1[u];
        for (int i = 0; i < din; ++i) a += xin[i] * W1[i * 128 + u];
        h[j] = a;
    }
    float s = 0.f, s2 = 0.f;
    #pragma unroll
    for (int j = 0; j < 4; ++j) { s += h[j]; s2 += h[j] * h[j]; }
    #pragma unroll
    for (int m = 1; m < 32; m <<= 1) { s += __shfl_xor(s, m, 32); s2 += __shfl_xor(s2, m, 32); }
    float mean = s * (1.f / 128.f);
    float var  = s2 * (1.f / 128.f) - mean * mean;
    float inv  = rsqrtf(var + 1e-5f);
    #pragma unroll
    for (int j = 0; j < 4; ++j) {
        int u = lane + 32 * j;
        float v = (h[j] - mean) * inv * G[u] + BE[u];
        hbuf[w][u] = v > 0.f ? v : 0.f;
    }
    WAVE_DS_FENCE();
    #pragma unroll
    for (int j = 0; j < 2; ++j) {
        int v = lane + 32 * j;
        float o = B2[v];
        for (int u = 0; u < 128; ++u) o += hbuf[w][u] * W2[u * 64 + v];
        xout[(size_t)n * 128 + v] = o;
    }
}

// ---------------------------------------------------------------------------
// agg[pj] += x[ti]  (only platform nodes receive messages)
// ---------------------------------------------------------------------------
__global__ void zero_kernel(float* __restrict__ p, int nfl) {
    int i = blockIdx.x * 256 + threadIdx.x;
    if (i < nfl) p[i] = 0.f;
}

__global__ void scatter_kernel(const float* __restrict__ xin, const int* __restrict__ ei,
                               float* __restrict__ agg, int dinSh) {
    int tid = blockIdx.x * 256 + threadIdx.x;
    int sh  = dinSh - 2;                       // float4 chunks per row
    int e   = tid >> sh;
    int cc  = (tid & ((1 << sh) - 1)) << 2;
    int t   = ei[e];
    int d   = ei[NE + e] - NT;
    float4 v = *(const float4*)(xin + (size_t)t * 128 + cc);
    float* a = agg + (size_t)d * 128 + cc;
    atomicAdd(a + 0, v.x); atomicAdd(a + 1, v.y);
    atomicAdd(a + 2, v.z); atomicAdd(a + 3, v.w);
}

// ---------------------------------------------------------------------------
// GIN layer: y = act2( W2^T * relu(W1^T * (x+agg) + b1) + b2 )
// Weights staged in LDS (block-shared). One wave per 16-node tile, wave-level
// tile loop. nt loop kept rolled (one live accumulator, no spills); per nt the
// KT B-fragments are burst-loaded to registers, then a KT-deep WMMA chain runs
// behind a single dscnt wait.
// ---------------------------------------------------------------------------
template<int DIN, int DH, int DOUT, bool RELU, bool WF16>
__global__ void gin_gemm_kernel(
    const float* __restrict__ xin, const float* __restrict__ agg,
    const _Float16* __restrict__ w1t, const float* __restrict__ b1,
    const _Float16* __restrict__ w2t, const float* __restrict__ b2,
    float* __restrict__ xout, _Float16* __restrict__ xhout)
{
    constexpr int DINP = DIN + LPAD;
    constexpr int DHP  = DH + LPAD;
    constexpr int KT1  = DIN / 32, NT1 = DH / 16;
    constexpr int KT2  = DH / 32,  NT2 = DOUT / 16;
    __shared__ _Float16 ldsW1[DH * DINP];
    __shared__ _Float16 ldsW2[DOUT * DHP];
    __shared__ _Float16 ldsA[4][16 * DINP];
    __shared__ _Float16 ldsH[4][16 * DHP];

    stage_w(w1t, ldsW1, DH,   DIN, threadIdx.x, 128);
    stage_w(w2t, ldsW2, DOUT, DH,  threadIdx.x, 128);
    __syncthreads();

    const int lane  = threadIdx.x & 31;
    const int w     = threadIdx.x >> 5;
    const int n15   = lane & 15;
    const int mbase = (lane >> 4) << 3;
    const int NTILES = (NN + 15) / 16;         // 1595
    const int nWaves = gridDim.x * 4;

    for (int tile = blockIdx.x * 4 + w; tile < NTILES; tile += nWaves) {
        const int row0 = tile * 16;
        WAVE_DS_FENCE();   // prior tile's LDS reads done before restaging
        // stage A = x (+agg for platform rows), fp32 -> f16, zero tail rows
        for (int i = lane; i < 16 * DIN; i += 32) {
            int r = i / DIN, c = i - r * DIN;
            int row = row0 + r;
            float v = 0.f;
            if (row < NN) {
                v = xin[(size_t)row * 128 + c];
                if (row >= NT) v += agg[(size_t)(row - NT) * 128 + c];
            }
            ldsA[w][r * DINP + c] = (_Float16)v;
        }
        WAVE_DS_FENCE();

        // ---- GEMM1 + ReLU -> ldsH ----
        {
            const _Float16* arow = &ldsA[w][n15 * DINP];
            v16h afr[KT1];
            #pragma unroll
            for (int kt = 0; kt < KT1; ++kt) afr[kt] = frag_a(arow, kt * 32, lane);
            #pragma unroll 1
            for (int nt = 0; nt < NT1; ++nt) {
                const _Float16* brow = ldsW1 + (nt * 16 + n15) * DINP;
                v16h bfr[KT1];
                #pragma unroll
                for (int kt = 0; kt < KT1; ++kt) bfr[kt] = frag_b(brow, kt * 32, lane);
                float bias = b1[nt * 16 + n15];
                v8f c;
                #pragma unroll
                for (int r = 0; r < 8; ++r) c[r] = bias;
                #pragma unroll
                for (int kt = 0; kt < KT1; ++kt) c = wmma_f16(afr[kt], bfr[kt], c);
                #pragma unroll
                for (int r = 0; r < 8; ++r) {
                    float v = c[r] > 0.f ? c[r] : 0.f;
                    ldsH[w][(mbase + r) * DHP + nt * 16 + n15] = (_Float16)v;
                }
            }
        }
        WAVE_DS_FENCE();

        // ---- GEMM2 (+optional ReLU) -> global ----
        {
            const _Float16* hrow = &ldsH[w][n15 * DHP];
            v16h hfr[KT2];
            #pragma unroll
            for (int kt = 0; kt < KT2; ++kt) hfr[kt] = frag_a(hrow, kt * 32, lane);
            #pragma unroll 1
            for (int nt = 0; nt < NT2; ++nt) {
                const _Float16* brow = ldsW2 + (nt * 16 + n15) * DHP;
                v16h bfr[KT2];
                #pragma unroll
                for (int kt = 0; kt < KT2; ++kt) bfr[kt] = frag_b(brow, kt * 32, lane);
                float bias = b2[nt * 16 + n15];
                v8f c;
                #pragma unroll
                for (int r = 0; r < 8; ++r) c[r] = bias;
                #pragma unroll
                for (int kt = 0; kt < KT2; ++kt) c = wmma_f16(hfr[kt], bfr[kt], c);
                #pragma unroll
                for (int r = 0; r < 8; ++r) {
                    int row = row0 + mbase + r;
                    if (row < NN) {
                        float v = c[r];
                        if (RELU) v = v > 0.f ? v : 0.f;
                        int col = nt * 16 + n15;
                        xout[(size_t)row * 128 + col] = v;
                        if (WF16) xhout[(size_t)row * 128 + col] = (_Float16)v;
                    }
                }
            }
        }
    }
}

// ---------------------------------------------------------------------------
// Edge MLP: gather [task|plat|ea] (K=131 pad 160) -> 128 relu -> dot w2 -> clip
// Weights staged in LDS; wave-level tile loop over 16-edge tiles; rolled nt
// loop with per-nt B-fragment burst; second layer fused as per-N-tile dot +
// 16-lane shuffle reduction.
// ---------------------------------------------------------------------------
__global__ void edge_kernel(
    const _Float16* __restrict__ xh, const float* __restrict__ ea,
    const int* __restrict__ ei,
    const _Float16* __restrict__ w1t, const float* __restrict__ b1,
    const float* __restrict__ w2,  const float* __restrict__ b2,
    float* __restrict__ out)
{
    __shared__ _Float16 ldsW[128 * KEP];       // 42KB
    __shared__ _Float16 ldsF[4][16 * KEP];     // 21KB

    stage_w(w1t, ldsW, 128, KE, threadIdx.x, 128);
    __syncthreads();

    const int lane  = threadIdx.x & 31;
    const int w     = threadIdx.x >> 5;
    const int n15   = lane & 15;
    const int mbase = (lane >> 4) << 3;
    const int NTILE_E = NE / 16;               // 50000
    const int nWaves  = gridDim.x * 4;

    for (int tile = blockIdx.x * 4 + w; tile < NTILE_E; tile += nWaves) {
        const int e0 = tile * 16;
        WAVE_DS_FENCE();
        // stage: 2 lanes per edge row
        {
            int row = lane >> 1;
            int e   = e0 + row;
            if ((lane & 1) == 0) {
                int t = ei[e];                               // task node row
                const int4* src = (const int4*)(xh + (size_t)t * 128);
                int4* dst = (int4*)&ldsF[w][row * KEP];
                #pragma unroll
                for (int j = 0; j < 8; ++j) dst[j] = src[j]; // 64 f16
            } else {
                int p = ei[NE + e];                          // NT+pj node row
                const int4* src = (const int4*)(xh + (size_t)p * 128);
                int4* dst = (int4*)&ldsF[w][row * KEP + 64];
                #pragma unroll
                for (int j = 0; j < 8; ++j) dst[j] = src[j]; // 64 f16
                #pragma unroll
                for (int c = 0; c < 32; ++c) {
                    float v = (c < 3) ? ea[(size_t)e * 3 + c] : 0.f;
                    ldsF[w][row * KEP + 128 + c] = (_Float16)v;
                }
            }
        }
        WAVE_DS_FENCE();

        const _Float16* arow = &ldsF[w][n15 * KEP];
        v16h afr[5];
        #pragma unroll
        for (int kt = 0; kt < 5; ++kt) afr[kt] = frag_a(arow, kt * 32, lane);

        float sacc[8];
        #pragma unroll
        for (int r = 0; r < 8; ++r) sacc[r] = 0.f;

        #pragma unroll 1
        for (int nt = 0; nt < 8; ++nt) {
            const _Float16* brow = ldsW + (nt * 16 + n15) * KEP;
            v16h bfr[5];
            #pragma unroll
            for (int kt = 0; kt < 5; ++kt) bfr[kt] = frag_b(brow, kt * 32, lane);
            float bias = b1[nt * 16 + n15];
            v8f c;
            #pragma unroll
            for (int r = 0; r < 8; ++r) c[r] = bias;
            #pragma unroll
            for (int kt = 0; kt < 5; ++kt) c = wmma_f16(afr[kt], bfr[kt], c);
            float w2v = w2[nt * 16 + n15];
            #pragma unroll
            for (int r = 0; r < 8; ++r) {
                float hval = c[r] > 0.f ? c[r] : 0.f;
                sacc[r] += hval * w2v;
            }
        }
        // reduce over N (16 lanes per half-wave)
        #pragma unroll
        for (int m = 1; m < 16; m <<= 1)
            #pragma unroll
            for (int r = 0; r < 8; ++r) sacc[r] += __shfl_xor(sacc[r], m, 32);

        if (n15 == 0) {
            float bb = b2[0];
            #pragma unroll
            for (int r = 0; r < 8; ++r) {
                float s = sacc[r] + bb;
                s = fminf(fmaxf(s, -50.f), 50.f);
                out[e0 + mbase + r] = s;
            }
        }
    }
}

// ---------------------------------------------------------------------------
// Launch
// ---------------------------------------------------------------------------
extern "C" void kernel_launch(void* const* d_in, const int* in_sizes, int n_in,
                              void* d_out, int out_size, void* d_ws, size_t ws_size,
                              hipStream_t stream) {
    (void)in_sizes; (void)n_in; (void)out_size; (void)ws_size;
    // JAX pytree leaf order (dict keys sorted):
    // 0 task_features  1 platform_features  2 edge_attr  3 edge_index
    // edge{l1.b,l1.w,l2.b,l2.w}=4..7, gin[0..2]{l1.b,l1.w,l2.b,l2.w}=8..19,
    // plat_enc{be,g,l1.b,l1.w,l2.b,l2.w}=20..25, task_enc{...}=26..31
    const float* tf = (const float*)d_in[0];
    const float* pf = (const float*)d_in[1];
    const float* ea = (const float*)d_in[2];
    const int*   ei = (const int*)  d_in[3];
    const float* e_b1 = (const float*)d_in[4];
    const float* e_w1 = (const float*)d_in[5];
    const float* e_b2 = (const float*)d_in[6];
    const float* e_w2 = (const float*)d_in[7];
    const float* g_b1[3] = {(const float*)d_in[8],  (const float*)d_in[12], (const float*)d_in[16]};
    const float* g_w1[3] = {(const float*)d_in[9],  (const float*)d_in[13], (const float*)d_in[17]};
    const float* g_b2[3] = {(const float*)d_in[10], (const float*)d_in[14], (const float*)d_in[18]};
    const float* g_w2[3] = {(const float*)d_in[11], (const float*)d_in[15], (const float*)d_in[19]};
    const float* p_be = (const float*)d_in[20]; const float* p_g  = (const float*)d_in[21];
    const float* p_b1 = (const float*)d_in[22]; const float* p_w1 = (const float*)d_in[23];
    const float* p_b2 = (const float*)d_in[24]; const float* p_w2 = (const float*)d_in[25];
    const float* t_be = (const float*)d_in[26]; const float* t_g  = (const float*)d_in[27];
    const float* t_b1 = (const float*)d_in[28]; const float* t_w1 = (const float*)d_in[29];
    const float* t_b2 = (const float*)d_in[30]; const float* t_w2 = (const float*)d_in[31];

    // workspace carve-up (~33.1 MB)
    float* xA = (float*)d_ws;
    float* xB = xA + (size_t)NN * 128;
    _Float16* xh  = (_Float16*)(xB + (size_t)NN * 128);
    float*    agg = (float*)(xh + (size_t)NN * 128);
    _Float16* g0w1t = (_Float16*)(agg + (size_t)NP * 128);
    _Float16* g0w2t = g0w1t + 128 * 64;
    _Float16* g1w1t = g0w2t + 128 * 128;
    _Float16* g1w2t = g1w1t + 128 * 128;
    _Float16* g2w1t = g1w2t + 128 * 128;
    _Float16* g2w2t = g2w1t + 64 * 128;
    _Float16* ew1t  = g2w2t + 64 * 64;

    // 1) weight prep (f16, transposed [N][K], packed)
    #define WPREP(src, dst, K, N, KP) \
        wprep_kernel<<<((N)*(KP) + 255) / 256, 256, 0, stream>>>(src, dst, K, N, KP)
    WPREP(g_w1[0], g0w1t,  64, 128,  64);
    WPREP(g_w2[0], g0w2t, 128, 128, 128);
    WPREP(g_w1[1], g1w1t, 128, 128, 128);
    WPREP(g_w2[1], g1w2t, 128, 128, 128);
    WPREP(g_w1[2], g2w1t, 128,  64, 128);
    WPREP(g_w2[2], g2w2t,  64,  64,  64);
    WPREP(e_w1,    ew1t,  131, 128, KE);
    #undef WPREP

    // 2) encoders -> xA (fp32, stride 128, cols 0..63)
    encoder_kernel<<<NN / 8, 256, 0, stream>>>(
        tf, pf, t_w1, t_b1, t_g, t_be, t_w2, t_b2,
                p_w1, p_b1, p_g, p_be, p_w2, p_b2, xA);

    // 3) GIN layers (ping-pong xA/xB); 200 blocks x 4 waves, ~2 tiles/wave
    const int ginBlocks = 200;
    // layer 0: 64 -> 128 -> 128, relu
    zero_kernel<<<(NP * 128) / 256, 256, 0, stream>>>(agg, NP * 128);
    scatter_kernel<<<(NE * 16) / 256, 256, 0, stream>>>(xA, ei, agg, 6);
    gin_gemm_kernel<64, 128, 128, true, false><<<ginBlocks, 128, 0, stream>>>(
        xA, agg, g0w1t, g_b1[0], g0w2t, g_b2[0], xB, (_Float16*)nullptr);
    // layer 1: 128 -> 128 -> 128, relu
    zero_kernel<<<(NP * 128) / 256, 256, 0, stream>>>(agg, NP * 128);
    scatter_kernel<<<(NE * 32) / 256, 256, 0, stream>>>(xB, ei, agg, 7);
    gin_gemm_kernel<128, 128, 128, true, false><<<ginBlocks, 128, 0, stream>>>(
        xB, agg, g1w1t, g_b1[1], g1w2t, g_b2[1], xA, (_Float16*)nullptr);
    // layer 2: 128 -> 64 -> 64, no relu; emit f16 mirror for edge gathers
    zero_kernel<<<(NP * 128) / 256, 256, 0, stream>>>(agg, NP * 128);
    scatter_kernel<<<(NE * 32) / 256, 256, 0, stream>>>(xA, ei, agg, 7);
    gin_gemm_kernel<128, 64, 64, false, true><<<ginBlocks, 128, 0, stream>>>(
        xA, agg, g2w1t, g_b1[2], g2w2t, g_b2[2], xB, xh);

    // 4) edge MLP -> scores; 2500 blocks x 4 waves, ~5 tiles/wave
    edge_kernel<<<2500, 128, 0, stream>>>(xh, ea, ei, ew1t, e_b1, e_w2, e_b2,
                                          (float*)d_out);
}
